// OrthoHydraLoRAExpModule_326417514833
// MI455X (gfx1250) — matrix operand order repaired and checked by reading, not compiled, not run on hardware
//
#include <hip/hip_runtime.h>
#include <math.h>

// ---------- problem constants ----------
#define BB 4
#define SS 2048
#define INF 4096
#define OUTF 4096
#define MM (BB*SS)          // 8192
#define RR 64
#define EE 8
#define SIGF 256
#define SIGH 128
#define KSCALE (1.0f/64.0f) // MULT * SCALE

typedef __attribute__((ext_vector_type(16))) __bf16 v16bf;
typedef __attribute__((ext_vector_type(8)))  __bf16 v8bf;
typedef __attribute__((ext_vector_type(8)))  float  v8f;

// ---------- helpers ----------
static __device__ inline __bf16 to_bf16(float f) {
    unsigned u = __builtin_bit_cast(unsigned, f);
    unsigned r = (u + 0x7FFFu + ((u >> 16) & 1u)) >> 16;
    unsigned short s = (unsigned short)r;
    return __builtin_bit_cast(__bf16, s);
}

static __device__ inline v8f zero8() {
    v8f z;
#pragma unroll
    for (int i = 0; i < 8; i++) z[i] = 0.f;
    return z;
}

// A-fragment (16-bit 16x32): lo chunk at +0, hi chunk at +16 elements
static __device__ inline v16bf load_afrag(const __bf16* base) {
    union { v16bf v; v8bf h[2]; } u;
    u.h[0] = *reinterpret_cast<const v8bf*>(base);
    u.h[1] = *reinterpret_cast<const v8bf*>(base + 16);
    return u.v;
}
// B-fragment (16-bit 32x16, one column): lo chunk at +0, hi chunk at +16 elements
static __device__ inline v16bf load_bfrag(const __bf16* base) {
    union { v16bf v; v8bf h[2]; } u;
    u.h[0] = *reinterpret_cast<const v8bf*>(base);
    u.h[1] = *reinterpret_cast<const v8bf*>(base + 16);
    return u.v;
}

static __device__ inline void load_a4(v16bf dst[4], const __bf16* const p[4], int k) {
#pragma unroll
    for (int f = 0; f < 4; f++) dst[f] = load_afrag(p[f] + k);
}
static __device__ inline void load_b4(v16bf dst[4], const __bf16* const p[4], int k) {
#pragma unroll
    for (int g = 0; g < 4; g++) dst[g] = load_bfrag(p[g] + k);
}
static __device__ inline void mma16(v8f acc[4][4], const v16bf a[4], const v16bf b[4]) {
#pragma unroll
    for (int f = 0; f < 4; f++)
#pragma unroll
        for (int g = 0; g < 4; g++)
            acc[f][g] = __builtin_amdgcn_wmma_f32_16x16x32_bf16(
                false, a[f], false, b[g], (short)0, acc[f][g], false, false);
}

// ---------- 1. f32 -> bf16 convert ----------
__global__ void cvt_bf16_kernel(const float* __restrict__ src,
                                __bf16* __restrict__ dst, int n) {
    int i = blockIdx.x * 256 + threadIdx.x;
    if (i < n) dst[i] = to_bf16(src[i]);
}

// ---------- 2. Cayley: R = (I+A)^-1 (I-A), A = S - S^T, 64x64 Gauss-Jordan ----------
__global__ void cayley_kernel(const float* __restrict__ Sq,
                              const float* __restrict__ Sp,
                              float* __restrict__ Rq,
                              float* __restrict__ Rp) {
    __shared__ float Mm[64 * 64];
    __shared__ float Rr[64 * 64];
    __shared__ float ck[64];
    const float* S; float* Ro;
    if (blockIdx.x == 0) { S = Sq; Ro = Rq; }
    else { S = Sp + (size_t)(blockIdx.x - 1) * 4096; Ro = Rp + (size_t)(blockIdx.x - 1) * 4096; }
    int j = threadIdx.x; // 64 threads, thread j owns column j
    for (int i = 0; i < 64; i++) {
        float a  = S[i * 64 + j] - S[j * 64 + i];
        float id = (i == j) ? 1.f : 0.f;
        Mm[i * 64 + j] = id + a;
        Rr[i * 64 + j] = id - a;
    }
    __syncthreads();
    for (int k = 0; k < 64; k++) {
        float pinv = 1.f / Mm[k * 64 + k];
        float mkj = Mm[k * 64 + j] * pinv;
        float rkj = Rr[k * 64 + j] * pinv;
        ck[j] = Mm[j * 64 + k];          // snapshot column k (pre-elimination)
        __syncthreads();
        Mm[k * 64 + j] = mkj;
        Rr[k * 64 + j] = rkj;
        for (int i = 0; i < 64; i++) {
            if (i == k) continue;
            float f = ck[i];
            Mm[i * 64 + j] -= f * mkj;
            Rr[i * 64 + j] -= f * rkj;
        }
        __syncthreads();
    }
    for (int i = 0; i < 64; i++) Ro[i * 64 + j] = Rr[i * 64 + j];
}

// ---------- 3. down'[r][i] = lam[r]*SCALE * sum_k Rq[r][k] Q[k][i]  (bf16) ----------
__global__ void down_kernel(const float* __restrict__ Rq,
                            const float* __restrict__ Q,
                            const float* __restrict__ lam,
                            __bf16* __restrict__ downb) {
    int gid = blockIdx.x * 256 + threadIdx.x;  // RR*INF threads
    int r = gid >> 12, i = gid & 4095;
    float acc = 0.f;
#pragma unroll 8
    for (int k = 0; k < 64; k++) acc += Rq[r * 64 + k] * Q[k * INF + i];
    downb[(size_t)r * INF + i] = to_bf16(acc * lam[r] * KSCALE);
}

// ---------- 4. up[e][o][r] = sum_k P[o][k] Rp[e][k][r] ----------
__global__ void up_kernel(const float* __restrict__ P,
                          const float* __restrict__ Rp,
                          float* __restrict__ up) {
    int gid = blockIdx.x * 256 + threadIdx.x;  // EE*OUTF*RR threads
    int e = gid >> 18;
    int rem = gid & 262143;
    int o = rem >> 6, r = rem & 63;
    const float* rp = Rp + (size_t)e * 4096;
    float acc = 0.f;
#pragma unroll 8
    for (int k = 0; k < 64; k++) acc += P[o * 64 + k] * rp[k * 64 + r];
    up[gid] = acc;
}

// ---------- 5. pooled[b][i] = mean_s x[b][s][i] ----------
__global__ void pooled_kernel(const float* __restrict__ x,
                              float* __restrict__ pooled) {
    int gid = blockIdx.x * 256 + threadIdx.x;  // BB*INF threads
    int b = gid >> 12, i = gid & 4095;
    const float* xp = x + (size_t)b * SS * INF + i;
    float acc = 0.f;
    for (int s = 0; s < SS; s++) acc += xp[(size_t)s * INF];
    pooled[gid] = acc * (1.0f / SS);
}

// ---------- 6. logits[b][e] = pooled[b].rW[e] + rb[e] ----------
__global__ void router_logits_kernel(const float* __restrict__ pooled,
                                     const float* __restrict__ rW,
                                     const float* __restrict__ rb,
                                     float* __restrict__ logits) {
    __shared__ float red[256];
    int b = blockIdx.x >> 3, e = blockIdx.x & 7;
    int t = threadIdx.x;
    float acc = 0.f;
    for (int i = t; i < INF; i += 256)
        acc += pooled[b * INF + i] * rW[(size_t)e * INF + i];
    red[t] = acc;
    __syncthreads();
    for (int s = 128; s > 0; s >>= 1) {
        if (t < s) red[t] += red[t + s];
        __syncthreads();
    }
    if (t == 0) logits[b * EE + e] = red[0] + rb[e];
}

// ---------- 7. sigma MLP + softmax gate ----------
__global__ void gate_kernel(const float* __restrict__ sigma,
                            const float* __restrict__ W1,
                            const float* __restrict__ b1,
                            const float* __restrict__ W2,
                            const float* __restrict__ logits,
                            float* __restrict__ gate) {
    __shared__ float h[BB][SIGH];
    __shared__ float lg[BB][EE];
    int t = threadIdx.x;  // 256
    const float nl10k = -logf(10000.f);
    for (int task = t; task < BB * SIGH; task += 256) {
        int b = task >> 7, j = task & 127;
        float sig = sigma[b];
        float acc = b1[j];
        for (int f = 0; f < SIGF; f++) {
            int idx = (f < 128) ? f : (f - 128);
            float freq = __expf(nl10k * (float)idx * (1.0f / 128.0f));
            float ang = sig * freq;
            float feat = (f < 128) ? __cosf(ang) : __sinf(ang);
            acc += feat * W1[j * SIGF + f];
        }
        h[b][j] = acc / (1.f + __expf(-acc));  // silu
    }
    __syncthreads();
    if (t < BB * EE) {
        int b = t >> 3, e = t & 7;
        float acc = logits[b * EE + e];
        for (int j = 0; j < SIGH; j++) acc += h[b][j] * W2[e * SIGH + j];
        lg[b][e] = acc;
    }
    __syncthreads();
    if (t < BB) {
        float mx = lg[t][0];
        for (int e = 1; e < EE; e++) mx = fmaxf(mx, lg[t][e]);
        float ex[EE], sum = 0.f;
        for (int e = 0; e < EE; e++) { ex[e] = __expf(lg[t][e] - mx); sum += ex[e]; }
        float inv = 1.f / sum;
        for (int e = 0; e < EE; e++) gate[t * EE + e] = ex[e] * inv;
    }
}

// ---------- 8. up_b[b][o][r] = sum_e gate[b][e] up[e][o][r]  (bf16) ----------
__global__ void upb_kernel(const float* __restrict__ gate,
                           const float* __restrict__ up,
                           __bf16* __restrict__ upbb) {
    int gid = blockIdx.x * 256 + threadIdx.x;  // BB*OUTF*RR threads
    int b = gid >> 18;
    int rem = gid & 262143;
    float acc = 0.f;
#pragma unroll
    for (int e = 0; e < EE; e++)
        acc += gate[b * EE + e] * up[(size_t)e * OUTF * RR + rem];
    upbb[gid] = to_bf16(acc);
}

// ---------- 9. lx = x @ down'^T  (WMMA, M=8192 N=64 K=4096) -> bf16 ----------
// register double-buffered so loads of step k+1 overlap WMMAs of step k
__global__ __launch_bounds__(256)
void lx_gemm_kernel(const __bf16* __restrict__ xb,
                    const __bf16* __restrict__ downb,
                    __bf16* __restrict__ lxb) {
    int lane = threadIdx.x & 31;
    int wid  = threadIdx.x >> 5;          // 8 waves, 16 M-rows each
    int col  = lane & 15;
    int aoff = (lane < 16) ? 0 : 8;
    int boff = (lane < 16) ? 0 : 8;
    int mrow = blockIdx.x * 128 + wid * 16;
    v8f acc[4];
#pragma unroll
    for (int g = 0; g < 4; g++) acc[g] = zero8();

    const __bf16* ap = xb + (size_t)(mrow + col) * INF + aoff;
    const __bf16* bp[4];
#pragma unroll
    for (int g = 0; g < 4; g++) bp[g] = downb + (size_t)(g * 16 + col) * INF + boff;

    v16bf a0, a1, b0[4], b1[4];
    a0 = load_afrag(ap);
    load_b4(b0, bp, 0);
    int k = 32;
    for (int it = 0; it < 63; ++it) {
        a1 = load_afrag(ap + k);
        load_b4(b1, bp, k);
#pragma unroll
        for (int g = 0; g < 4; g++)
            acc[g] = __builtin_amdgcn_wmma_f32_16x16x32_bf16(
                false, a0, false, b0[g], (short)0, acc[g], false, false);
        k += 32;
        a0 = load_afrag(ap + k);
        load_b4(b0, bp, k);
#pragma unroll
        for (int g = 0; g < 4; g++)
            acc[g] = __builtin_amdgcn_wmma_f32_16x16x32_bf16(
                false, a1, false, b1[g], (short)0, acc[g], false, false);
        k += 32;
    }
    // k == 4064 here: last step
    a1 = load_afrag(ap + k);
    load_b4(b1, bp, k);
#pragma unroll
    for (int g = 0; g < 4; g++)
        acc[g] = __builtin_amdgcn_wmma_f32_16x16x32_bf16(
            false, a0, false, b0[g], (short)0, acc[g], false, false);
#pragma unroll
    for (int g = 0; g < 4; g++)
        acc[g] = __builtin_amdgcn_wmma_f32_16x16x32_bf16(
            false, a1, false, b1[g], (short)0, acc[g], false, false);

    int r0 = mrow + ((lane < 16) ? 0 : 8);
#pragma unroll
    for (int g = 0; g < 4; g++)
#pragma unroll
        for (int v = 0; v < 8; v++)
            lxb[(size_t)(r0 + v) * RR + g * 16 + col] = to_bf16(acc[g][v]);
}

// ---------- 10. fused main GEMM: out = x@W^T + bias + lx@up_b[b]^T ----------
// block tile 128x256; 8 waves as 2(M) x 4(N); wave tile 64x64 (16 WMMA accums)
// register double-buffered K pipeline
__global__ __launch_bounds__(256)
void main_gemm_kernel(const __bf16* __restrict__ xb,
                      const __bf16* __restrict__ Wb,
                      const float*  __restrict__ bias,
                      const __bf16* __restrict__ lxb,
                      const __bf16* __restrict__ upbb,
                      float* __restrict__ out) {
    int lane = threadIdx.x & 31;
    int wid  = threadIdx.x >> 5;
    int wm = wid & 1, wn = wid >> 1;
    int mrow0 = blockIdx.y * 128 + wm * 64;
    int ncol0 = blockIdx.x * 256 + wn * 64;
    int batch = blockIdx.y >> 4;          // 16 M-tiles of 128 per batch of 2048
    int col  = lane & 15;
    int aoff = (lane < 16) ? 0 : 8;
    int boff = (lane < 16) ? 0 : 8;

    v8f acc[4][4];
#pragma unroll
    for (int f = 0; f < 4; f++)
#pragma unroll
        for (int g = 0; g < 4; g++) acc[f][g] = zero8();

    const __bf16* aptr[4];
    const __bf16* bptr[4];
#pragma unroll
    for (int f = 0; f < 4; f++)
        aptr[f] = xb + (size_t)(mrow0 + f * 16 + col) * INF + aoff;
#pragma unroll
    for (int g = 0; g < 4; g++)
        bptr[g] = Wb + (size_t)(ncol0 + g * 16 + col) * INF + boff;

    v16bf a0[4], b0[4], a1[4], b1[4];
    // prologue: step 0
    load_a4(a0, aptr, 0);
    load_b4(b0, bptr, 0);
    int k = 32;
    // 63 double-iterations: consume steps 0..125, load steps 1..126
    for (int it = 0; it < 63; ++it) {
        load_a4(a1, aptr, k);
        load_b4(b1, bptr, k);
        mma16(acc, a0, b0);
        k += 32;
        load_a4(a0, aptr, k);
        load_b4(b0, bptr, k);
        mma16(acc, a1, b1);
        k += 32;
    }
    // tail: k == 4064 (step 127)
    load_a4(a1, aptr, k);
    load_b4(b1, bptr, k);
    mma16(acc, a0, b0);   // step 126
    mma16(acc, a1, b1);   // step 127

    // LoRA extension: two K=32 steps over [lx | up_b[batch]]
    {
        const __bf16* ub = upbb + (size_t)batch * OUTF * RR;
        const __bf16* laptr[4];
        const __bf16* lbptr[4];
#pragma unroll
        for (int f = 0; f < 4; f++)
            laptr[f] = lxb + (size_t)(mrow0 + f * 16 + col) * RR + aoff;
#pragma unroll
        for (int g = 0; g < 4; g++)
            lbptr[g] = ub + (size_t)(ncol0 + g * 16 + col) * RR + boff;
        load_a4(a0, laptr, 0);
        load_b4(b0, lbptr, 0);
        load_a4(a1, laptr, 32);
        load_b4(b1, lbptr, 32);
        mma16(acc, a0, b0);
        mma16(acc, a1, b1);
    }

    // epilogue: + bias, store f32
    int rhalf = (lane < 16) ? 0 : 8;
#pragma unroll
    for (int g = 0; g < 4; g++) {
        int ncolg = ncol0 + g * 16 + col;
        float bv = bias[ncolg];
#pragma unroll
        for (int f = 0; f < 4; f++) {
            int r0 = mrow0 + f * 16 + rhalf;
#pragma unroll
            for (int v = 0; v < 8; v++)
                out[(size_t)(r0 + v) * OUTF + ncolg] = acc[f][g][v] + bv;
        }
    }
}

// ---------- launch ----------
extern "C" void kernel_launch(void* const* d_in, const int* in_sizes, int n_in,
                              void* d_out, int out_size, void* d_ws, size_t ws_size,
                              hipStream_t stream) {
    const float* x        = (const float*)d_in[0];
    const float* W        = (const float*)d_in[1];
    const float* bias     = (const float*)d_in[2];
    const float* P_basis  = (const float*)d_in[3];
    const float* Q_basis  = (const float*)d_in[4];
    const float* S_q      = (const float*)d_in[5];
    const float* S_p      = (const float*)d_in[6];
    const float* lam      = (const float*)d_in[7];
    const float* router_W = (const float*)d_in[8];
    const float* router_b = (const float*)d_in[9];
    const float* sig_W1   = (const float*)d_in[10];
    const float* sig_b1   = (const float*)d_in[11];
    const float* sig_W2   = (const float*)d_in[12];
    const float* sigma    = (const float*)d_in[13];
    float* out = (float*)d_out;

    char* ws = (char*)d_ws;
    __bf16* XB    = (__bf16*)(ws + 0);            // 8192*4096*2  = 67108864
    __bf16* WB    = (__bf16*)(ws + 67108864);     // 4096*4096*2  = 33554432
    float*  RQ    = (float*) (ws + 100663296);    // 64*64*4      = 16384
    float*  RP    = (float*) (ws + 100679680);    // 8*64*64*4    = 131072
    __bf16* DOWNB = (__bf16*)(ws + 100810752);    // 64*4096*2    = 524288
    float*  UP    = (float*) (ws + 101335040);    // 8*4096*64*4  = 8388608
    float*  POOLED= (float*) (ws + 109723648);    // 4*4096*4     = 65536
    float*  LOGITS= (float*) (ws + 109789184);    // 256
    float*  GATE  = (float*) (ws + 109789440);    // 256
    __bf16* UPBB  = (__bf16*)(ws + 109789696);    // 4*4096*64*2  = 2097152
    __bf16* LXB   = (__bf16*)(ws + 111886848);    // 8192*64*2    = 1048576

    // 1. convert x, W to bf16
    cvt_bf16_kernel<<<(MM * INF + 255) / 256, 256, 0, stream>>>(x, XB, MM * INF);
    cvt_bf16_kernel<<<(OUTF * INF + 255) / 256, 256, 0, stream>>>(W, WB, OUTF * INF);
    // 2. Cayley rotations (1 for S_q + 8 for S_p)
    cayley_kernel<<<1 + EE, 64, 0, stream>>>(S_q, S_p, RQ, RP);
    // 3. down' (lam & scale folded), bf16
    down_kernel<<<(RR * INF) / 256, 256, 0, stream>>>(RQ, Q_basis, lam, DOWNB);
    // 4. per-expert up projections
    up_kernel<<<(EE * OUTF * RR) / 256, 256, 0, stream>>>(P_basis, RP, UP);
    // 5. lx via WMMA -> bf16
    lx_gemm_kernel<<<MM / 128, 256, 0, stream>>>(XB, DOWNB, LXB);
    // 6-8. router gate
    pooled_kernel<<<(BB * INF) / 256, 256, 0, stream>>>(x, POOLED);
    router_logits_kernel<<<BB * EE, 256, 0, stream>>>(POOLED, router_W, router_b, LOGITS);
    gate_kernel<<<1, 256, 0, stream>>>(sigma, sig_W1, sig_b1, sig_W2, LOGITS, GATE);
    upb_kernel<<<(BB * OUTF * RR) / 256, 256, 0, stream>>>(GATE, UP, UPBB);
    // 9. fused main GEMM + LoRA + bias
    dim3 grid(OUTF / 256, MM / 128);
    main_gemm_kernel<<<grid, 256, 0, stream>>>(XB, WB, bias, LXB, UPBB, out);
}
